// Attention_14499809591345
// MI455X (gfx1250) — matrix operand order repaired and checked by reading, not compiled, not run on hardware
//
#include <hip/hip_runtime.h>
#include <hip/hip_fp16.h>

typedef __attribute__((ext_vector_type(16))) _Float16 v16h;
typedef __attribute__((ext_vector_type(8)))  _Float16 v8h;
typedef __attribute__((ext_vector_type(8)))  float    v8f;

#define EMBED_DIM 1024
#define NUM_HEADS 16
#define HEAD_SIZE 64
#define BATCH     4
#define SEQ       2048

#define LS 72   // LDS row stride (halfs): 144B rows, 16B-aligned 32-half chunks

// ---------------------------------------------------------------------------
// CDNA5 async global->LDS copy (GLOBAL_LOAD_ASYNC_TO_LDS_B128, ASYNCcnt).
// INST_OFFSET applies to both the LDS destination and the global source.
// ---------------------------------------------------------------------------
#define ASYNC_LDS_CP16(ldsaddr, gptr, OFF)                                    \
    asm volatile("global_load_async_to_lds_b128 %0, %1, off offset:" #OFF     \
                 :: "v"(ldsaddr), "v"(gptr) : "memory")

#define ASYNC_LDS_CP64(ldsaddr, gptr)                                         \
    do {                                                                      \
        ASYNC_LDS_CP16(ldsaddr, gptr, 0);                                     \
        ASYNC_LDS_CP16(ldsaddr, gptr, 16);                                    \
        ASYNC_LDS_CP16(ldsaddr, gptr, 32);                                    \
        ASYNC_LDS_CP16(ldsaddr, gptr, 48);                                    \
    } while (0)

#define WAIT_ASYNC() asm volatile("s_wait_asynccnt 0x0" ::: "memory")

static __device__ __forceinline__ uint32_t lds_addr_of(const void* p) {
    // generic->LDS aperture: addr[31:0] is the LDS byte address
    return (uint32_t)(uintptr_t)p;
}

// ---------------------------------------------------------------------------
// WMMA fragment helpers (wave32, 16x16x32 f16, CDNA5 ISA 7.12.2 layouts)
// A 16x32: lane(0-15)=row M; halfs 0-7 = K klo..klo+7, halfs 8-15 =
//          K klo+16..klo+23, klo = 8*(lane>=16).
// B 32x16: lane(0-15)=col N; halfs 0-7 = K klo..klo+7, halfs 8-15 =
//          K klo+16..klo+23, klo = 8*(lane>=16).
// C/D 16x16: vgpr r <-> M = r + 8*(lane>=16), N = lane&15.
// ---------------------------------------------------------------------------
static __device__ __forceinline__ v16h frag_a_row(const _Float16* row, int hi) {
    const _Float16* q = row + hi * 8;
    v8h lo = *(const v8h*)(q);
    v8h hh = *(const v8h*)(q + 16);
    return __builtin_shufflevector(lo, hh, 0,1,2,3,4,5,6,7,8,9,10,11,12,13,14,15);
}
static __device__ __forceinline__ v16h frag_b_row(const _Float16* row, int hi) {
    const _Float16* q = row + hi * 8;
    v8h lo = *(const v8h*)(q);
    v8h hh = *(const v8h*)(q + 16);
    return __builtin_shufflevector(lo, hh, 0,1,2,3,4,5,6,7,8,9,10,11,12,13,14,15);
}
static __device__ __forceinline__ v8f wmma_f16(v16h a, v16h b, v8f c) {
    return __builtin_amdgcn_wmma_f32_16x16x32_f16(false, a, false, b,
                                                  (short)0, c, false, false);
}
static __device__ __forceinline__ float redmax16(float v) {
    #pragma unroll
    for (int m = 1; m < 16; m <<= 1) v = fmaxf(v, __shfl_xor(v, m, 32));
    return v;
}
static __device__ __forceinline__ float redsum16(float v) {
    #pragma unroll
    for (int m = 1; m < 16; m <<= 1) v += __shfl_xor(v, m, 32);
    return v;
}

// ---------------------------------------------------------------------------
// f32 -> f16 flat convert (x)
// ---------------------------------------------------------------------------
__global__ void cvt_f32_f16(const float* __restrict__ src,
                            _Float16* __restrict__ dst, int n) {
    int i = blockIdx.x * blockDim.x + threadIdx.x;
    int stride = gridDim.x * blockDim.x;
    for (; i < n; i += stride) dst[i] = (_Float16)src[i];
}

// ---------------------------------------------------------------------------
// f32 -> f16 transpose-convert: dst[n*K + k] = src[k*N + n]   (weights)
// ---------------------------------------------------------------------------
__global__ __launch_bounds__(256)
void cvt_transpose_f32_f16(const float* __restrict__ src,
                           _Float16* __restrict__ dst, int K, int N) {
    __shared__ float tile[32][33];
    const int n0 = blockIdx.x * 32, k0 = blockIdx.y * 32;
    const int tx = threadIdx.x, ty = threadIdx.y;     // 32 x 8
    #pragma unroll
    for (int i = 0; i < 32; i += 8)
        tile[ty + i][tx] = src[(size_t)(k0 + ty + i) * N + n0 + tx];
    __syncthreads();
    #pragma unroll
    for (int i = 0; i < 32; i += 8)
        dst[(size_t)(n0 + ty + i) * K + k0 + tx] = (_Float16)tile[tx][ty + i];
}

// ---------------------------------------------------------------------------
// 64x64 GEMM tile engine: K-step 64, double-buffered async LDS staging.
// 128 threads = 4 waves; wave w computes rows [w*16,w*16+16) x 64 cols.
// A row-major [M][K]; Bt row-major [N][K] (pre-transposed weights).
// ---------------------------------------------------------------------------
static __device__ __forceinline__ void gemm_tile_64x64(
    const _Float16* __restrict__ A, int lda,
    const _Float16* __restrict__ Bt, int ldb,
    int K, int m0, int n0,
    _Float16* As, _Float16* Bs,          // each 2 * 64*LS halfs
    v8f acc[4])
{
    const int t = threadIdx.x, lane = t & 31, w = t >> 5;
    const int ml = lane & 15, hi = lane >> 4;
    #pragma unroll
    for (int nt = 0; nt < 4; ++nt) acc[nt] = (v8f){0,0,0,0,0,0,0,0};

    const int row = t >> 1, seg = t & 1;   // 64 rows x 2 segments of 32 halfs
    uint32_t lds_a[2], lds_b[2];
    #pragma unroll
    for (int u = 0; u < 2; ++u) {
        lds_a[u] = lds_addr_of(&As[u * 64 * LS + row * LS + seg * 32]);
        lds_b[u] = lds_addr_of(&Bs[u * 64 * LS + row * LS + seg * 32]);
    }

    auto stage = [&](int k0, int bufi) {
        const _Float16* asrc = A  + (size_t)(m0 + row) * lda + k0 + seg * 32;
        const _Float16* bsrc = Bt + (size_t)(n0 + row) * ldb + k0 + seg * 32;
        ASYNC_LDS_CP64(lds_a[bufi], asrc);
        ASYNC_LDS_CP64(lds_b[bufi], bsrc);
        __builtin_prefetch(asrc + 64, 0, 3);   // warm tile after next in L2
        __builtin_prefetch(bsrc + 64, 0, 3);
    };

    stage(0, 0);
    int cur = 0;
    for (int k0 = 0; k0 < K; k0 += 64, cur ^= 1) {
        WAIT_ASYNC();          // our copies for buf[cur] done
        __syncthreads();       // everyone's copies done; buf[cur^1] free to refill
        if (k0 + 64 < K) stage(k0 + 64, cur ^ 1);   // overlaps with compute below

        const _Float16* Ab = As + cur * 64 * LS;
        const _Float16* Bb = Bs + cur * 64 * LS;
        #pragma unroll
        for (int kk = 0; kk < 2; ++kk) {
            v16h a  = frag_a_row(&Ab[(w * 16 + ml) * LS + kk * 32], hi);
            v16h b0 = frag_b_row(&Bb[(0 * 16 + ml) * LS + kk * 32], hi);
            v16h b1 = frag_b_row(&Bb[(1 * 16 + ml) * LS + kk * 32], hi);
            v16h b2 = frag_b_row(&Bb[(2 * 16 + ml) * LS + kk * 32], hi);
            v16h b3 = frag_b_row(&Bb[(3 * 16 + ml) * LS + kk * 32], hi);
            acc[0] = wmma_f16(a, b0, acc[0]);
            acc[1] = wmma_f16(a, b1, acc[1]);
            acc[2] = wmma_f16(a, b2, acc[2]);
            acc[3] = wmma_f16(a, b3, acc[3]);
        }
    }
}

// ---------------------------------------------------------------------------
// Kernel 1: QKV = x @ w_attn^T(pre-transposed) + b_attn
//   Q,K -> [B,H,S,Hd] f16 ;  V -> [B,H,Hd,S] f16 (transposed for P@V)
// ---------------------------------------------------------------------------
__global__ __launch_bounds__(128)
void qkv_gemm_kernel(const _Float16* __restrict__ xh,
                     const _Float16* __restrict__ wht,
                     const float* __restrict__ bias,
                     _Float16* __restrict__ qh,
                     _Float16* __restrict__ kh,
                     _Float16* __restrict__ vht)
{
    __shared__ _Float16 As[2 * 64 * LS];
    __shared__ _Float16 Bs[2 * 64 * LS];
    const int m0 = blockIdx.x * 64;
    const int n0 = blockIdx.y * 64;
    v8f acc[4];
    gemm_tile_64x64(xh, EMBED_DIM, wht, EMBED_DIM, EMBED_DIM, m0, n0, As, Bs, acc);

    const int lane = threadIdx.x & 31, w = threadIdx.x >> 5;
    const int ml = lane & 15, hi = lane >> 4;

    #pragma unroll
    for (int nt = 0; nt < 4; ++nt) {
        const int n     = n0 + nt * 16 + ml;
        const int which = n / EMBED_DIM;
        const int nh    = n % EMBED_DIM;
        const int h     = nh / HEAD_SIZE;
        const int d     = nh % HEAD_SIZE;
        const float bv  = bias[n];
        #pragma unroll
        for (int r = 0; r < 8; ++r) {
            const int m = m0 + w * 16 + r + hi * 8;
            const int b = m / SEQ, s = m % SEQ;
            const size_t bh = (size_t)b * NUM_HEADS + h;
            const _Float16 val = (_Float16)(acc[nt][r] + bv);
            if (which == 0)      qh [(bh * SEQ + s) * HEAD_SIZE + d] = val;
            else if (which == 1) kh [(bh * SEQ + s) * HEAD_SIZE + d] = val;
            else                 vht[(bh * HEAD_SIZE + d) * SEQ + s] = val;
        }
    }
}

// ---------------------------------------------------------------------------
// Kernel 2: causal flash attention. 128 threads = 4 waves; wave owns 16
// query rows (block = 64 queries of one (b,h)); 64-wide key tiles,
// double-buffered async K/V staging.
// ---------------------------------------------------------------------------
__global__ __launch_bounds__(128)
void flash_attn_kernel(const _Float16* __restrict__ Q,
                       const _Float16* __restrict__ K,
                       const _Float16* __restrict__ Vt,
                       _Float16* __restrict__ Y)
{
    __shared__ _Float16 Qs[64 * LS];
    __shared__ _Float16 Ks[2 * 64 * LS];    // [key][d]
    __shared__ _Float16 Vs[2 * 64 * LS];    // [d][key]   (from transposed V)
    __shared__ _Float16 Ps[4 * 16 * LS];    // per-wave P tile [16][64]

    const int bh = blockIdx.x;
    const int q0 = blockIdx.y * 64;
    const int b  = bh / NUM_HEADS;
    const int h  = bh % NUM_HEADS;
    const int t = threadIdx.x, lane = t & 31, w = t >> 5;
    const int ml = lane & 15, hi = lane >> 4;

    const _Float16* Qp  = Q  + (size_t)bh * SEQ * HEAD_SIZE;
    const _Float16* Kp  = K  + (size_t)bh * SEQ * HEAD_SIZE;
    const _Float16* Vpt = Vt + (size_t)bh * SEQ * HEAD_SIZE;   // [64][SEQ]

    const int row = t >> 1, seg = t & 1;
    const uint32_t lds_q = lds_addr_of(&Qs[row * LS + seg * 32]);
    uint32_t lds_k[2], lds_v[2];
    #pragma unroll
    for (int u = 0; u < 2; ++u) {
        lds_k[u] = lds_addr_of(&Ks[u * 64 * LS + row * LS + seg * 32]);
        lds_v[u] = lds_addr_of(&Vs[u * 64 * LS + row * LS + seg * 32]);
    }

    auto stageKV = [&](int kt2, int bufi) {
        const _Float16* ks = Kp  + (size_t)(kt2 + row) * HEAD_SIZE + seg * 32;
        const _Float16* vs = Vpt + (size_t)row * SEQ + kt2 + seg * 32;
        ASYNC_LDS_CP64(lds_k[bufi], ks);
        ASYNC_LDS_CP64(lds_v[bufi], vs);
        __builtin_prefetch(ks + 64 * HEAD_SIZE, 0, 3);
        __builtin_prefetch(vs + 64, 0, 3);
    };

    // stage Q tile + first K/V tile under one wait
    {
        const _Float16* src = Qp + (size_t)(q0 + row) * HEAD_SIZE + seg * 32;
        ASYNC_LDS_CP64(lds_q, src);
    }
    stageKV(0, 0);
    WAIT_ASYNC();
    __syncthreads();

    v16h qf[2];
    #pragma unroll
    for (int kk = 0; kk < 2; ++kk)
        qf[kk] = frag_a_row(&Qs[(w * 16 + ml) * LS + kk * 32], hi);

    v8f o[4];
    #pragma unroll
    for (int nt = 0; nt < 4; ++nt) o[nt] = (v8f){0,0,0,0,0,0,0,0};
    float rm[8], rl[8];
    #pragma unroll
    for (int r = 0; r < 8; ++r) { rm[r] = -1e30f; rl[r] = 0.0f; }

    _Float16* Pw = &Ps[w * 16 * LS];
    const int kend = q0 + 64;

    int cur = 0;
    for (int kt = 0; kt < kend; kt += 64, cur ^= 1) {
        // kick off next tile's copies; they overlap scores/softmax/PV below
        if (kt + 64 < kend) stageKV(kt + 64, cur ^ 1);

        const _Float16* Kb = Ks + cur * 64 * LS;
        const _Float16* Vb = Vs + cur * 64 * LS;

        // S = Q @ K^T : four 16x16 score tiles, Hd=64 -> 2 k-steps each
        v8f sc[4];
        #pragma unroll
        for (int j = 0; j < 4; ++j) {
            v16h bf0 = frag_b_row(&Kb[(j * 16 + ml) * LS + 0 * 32], hi);
            v16h bf1 = frag_b_row(&Kb[(j * 16 + ml) * LS + 1 * 32], hi);
            sc[j] = wmma_f16(qf[0], bf0, (v8f){0,0,0,0,0,0,0,0});
            sc[j] = wmma_f16(qf[1], bf1, sc[j]);
        }

        const bool diag = (kt + 64 > q0);   // only the diagonal tile needs a mask
        #pragma unroll
        for (int r = 0; r < 8; ++r) {
            float s[4];
            #pragma unroll
            for (int j = 0; j < 4; ++j) s[j] = sc[j][r] * 0.125f;
            if (diag) {
                const int qg = q0 + w * 16 + r + hi * 8;
                #pragma unroll
                for (int j = 0; j < 4; ++j)
                    if (kt + j * 16 + ml > qg) s[j] = -1e30f;
            }
            const float cmax  = fmaxf(fmaxf(s[0], s[1]), fmaxf(s[2], s[3]));
            const float mnew  = fmaxf(rm[r], redmax16(cmax));
            const float alpha = __expf(rm[r] - mnew);
            float psum = 0.0f;
            #pragma unroll
            for (int j = 0; j < 4; ++j) {
                const float p = __expf(s[j] - mnew);
                psum += p;
                Pw[(r + hi * 8) * LS + j * 16 + ml] = (_Float16)p;
            }
            rl[r] = rl[r] * alpha + redsum16(psum);
            rm[r] = mnew;
            #pragma unroll
            for (int nt = 0; nt < 4; ++nt) o[nt][r] *= alpha;
        }
        __syncthreads();   // P visible; all waves aligned before PV

        // O += P @ V : K-dim = 64 keys -> 2 A-fragments; V fragments contiguous
        #pragma unroll
        for (int kk = 0; kk < 2; ++kk) {
            v16h pf = frag_a_row(&Pw[ml * LS + kk * 32], hi);
            v16h b0 = frag_b_row(&Vb[(0 * 16 + ml) * LS + kk * 32], hi);
            v16h b1 = frag_b_row(&Vb[(1 * 16 + ml) * LS + kk * 32], hi);
            v16h b2 = frag_b_row(&Vb[(2 * 16 + ml) * LS + kk * 32], hi);
            v16h b3 = frag_b_row(&Vb[(3 * 16 + ml) * LS + kk * 32], hi);
            o[0] = wmma_f16(pf, b0, o[0]);
            o[1] = wmma_f16(pf, b1, o[1]);
            o[2] = wmma_f16(pf, b2, o[2]);
            o[3] = wmma_f16(pf, b3, o[3]);
        }

        WAIT_ASYNC();      // next tile's copies complete
        __syncthreads();   // all waves done with buf[cur] (PV) and Ps
    }

    // epilogue: normalize, write y in [B,S,D] f16
    #pragma unroll
    for (int nt = 0; nt < 4; ++nt) {
        #pragma unroll
        for (int r = 0; r < 8; ++r) {
            const int s   = q0 + w * 16 + r + hi * 8;
            const int col = h * HEAD_SIZE + nt * 16 + ml;
            Y[((size_t)b * SEQ + s) * EMBED_DIM + col] = (_Float16)(o[nt][r] / rl[r]);
        }
    }
}

// ---------------------------------------------------------------------------
// Kernel 3: out = y @ w_proj^T(pre-transposed) + b_proj (f32 out)
// ---------------------------------------------------------------------------
__global__ __launch_bounds__(128)
void proj_gemm_kernel(const _Float16* __restrict__ yh,
                      const _Float16* __restrict__ wht,
                      const float* __restrict__ bias,
                      float* __restrict__ out)
{
    __shared__ _Float16 As[2 * 64 * LS];
    __shared__ _Float16 Bs[2 * 64 * LS];
    const int m0 = blockIdx.x * 64;
    const int n0 = blockIdx.y * 64;
    v8f acc[4];
    gemm_tile_64x64(yh, EMBED_DIM, wht, EMBED_DIM, EMBED_DIM, m0, n0, As, Bs, acc);

    const int lane = threadIdx.x & 31, w = threadIdx.x >> 5;
    const int ml = lane & 15, hi = lane >> 4;

    #pragma unroll
    for (int nt = 0; nt < 4; ++nt) {
        const int n  = n0 + nt * 16 + ml;
        const float bv = bias[n];
        #pragma unroll
        for (int r = 0; r < 8; ++r) {
            const int m = m0 + w * 16 + r + hi * 8;
            out[(size_t)m * EMBED_DIM + n] = acc[nt][r] + bv;
        }
    }
}

// ---------------------------------------------------------------------------
// Launch
// ---------------------------------------------------------------------------
extern "C" void kernel_launch(void* const* d_in, const int* in_sizes, int n_in,
                              void* d_out, int out_size, void* d_ws, size_t ws_size,
                              hipStream_t stream)
{
    const float* x      = (const float*)d_in[0];
    const float* w_attn = (const float*)d_in[1];
    const float* b_attn = (const float*)d_in[2];
    const float* w_proj = (const float*)d_in[3];
    const float* b_proj = (const float*)d_in[4];
    float* out = (float*)d_out;

    const size_t nx  = (size_t)BATCH * SEQ * EMBED_DIM;       // 8,388,608
    const size_t nwa = (size_t)EMBED_DIM * 3 * EMBED_DIM;     // 3,145,728
    const size_t nwp = (size_t)EMBED_DIM * EMBED_DIM;         // 1,048,576

    char* ws = (char*)d_ws;
    _Float16* xh   = (_Float16*)ws; ws += nx  * sizeof(_Float16);
    _Float16* waht = (_Float16*)ws; ws += nwa * sizeof(_Float16);  // [3D][D]
    _Float16* wpht = (_Float16*)ws; ws += nwp * sizeof(_Float16);  // [D][D]
    _Float16* qh   = (_Float16*)ws; ws += nx  * sizeof(_Float16);  // [B,H,S,Hd]
    _Float16* kh   = (_Float16*)ws; ws += nx  * sizeof(_Float16);  // [B,H,S,Hd]
    _Float16* vht  = (_Float16*)ws; ws += nx  * sizeof(_Float16);  // [B,H,Hd,S]
    _Float16* yh   = (_Float16*)ws; ws += nx  * sizeof(_Float16);  // [B,S,D]

    cvt_f32_f16<<<4096, 256, 0, stream>>>(x, xh, (int)nx);
    cvt_transpose_f32_f16<<<dim3(3 * EMBED_DIM / 32, EMBED_DIM / 32),
                            dim3(32, 8), 0, stream>>>(w_attn, waht, EMBED_DIM, 3 * EMBED_DIM);
    cvt_transpose_f32_f16<<<dim3(EMBED_DIM / 32, EMBED_DIM / 32),
                            dim3(32, 8), 0, stream>>>(w_proj, wpht, EMBED_DIM, EMBED_DIM);

    // QKV: M=8192, N=3072, K=1024
    qkv_gemm_kernel<<<dim3((BATCH * SEQ) / 64, (3 * EMBED_DIM) / 64), 128, 0, stream>>>(
        xh, waht, b_attn, qh, kh, vht);

    // Flash attention: one block per (b*h, 64-query tile)
    flash_attn_kernel<<<dim3(BATCH * NUM_HEADS, SEQ / 64), 128, 0, stream>>>(
        qh, kh, vht, yh);

    // Projection: M=8192, N=1024, K=1024
    proj_gemm_kernel<<<dim3((BATCH * SEQ) / 64, EMBED_DIM / 64), 128, 0, stream>>>(
        yh, wpht, b_proj, out);
}
